// VFELayerMinusSlim_13417477833521
// MI455X (gfx1250) — matrix operand order repaired and checked by reading, not compiled, not run on hardware
//
#include <hip/hip_runtime.h>

typedef __attribute__((ext_vector_type(16))) _Float16 v16h;
typedef __attribute__((ext_vector_type(8)))  float    v8f;
typedef __attribute__((ext_vector_type(4)))  unsigned int v4u;

#define IN_CH  64
#define UNITS  128
#define NVOX   20000

// ---------------------------------------------------------------------------
// Kernel 1: zero the segment-max workspace (20000 x 128 u32). x >= 0 after the
// double-ReLU product, so 0 is the identity for max, and empty voxels are
// never gathered, so this matches the reference semantics.
// ---------------------------------------------------------------------------
__global__ __launch_bounds__(256) void k_seg_init(v4u* __restrict__ seg4, int n4) {
    int i = blockIdx.x * blockDim.x + threadIdx.x;
    if (i < n4) seg4[i] = (v4u){0u, 0u, 0u, 0u};
}

// ---------------------------------------------------------------------------
// Kernel 2: fused  x = relu(in@Wl^T+bl) * relu(mean@Ww^T+bw)
//           out[:, 0:128] = x ;  seg[idx[row]] = max(seg, x)   (u32 atomic)
// One 16-row tile per 128-thread block; wave w covers columns [32w, 32w+32).
// All hot addressing is unsigned 32-bit byte offsets off uniform bases so the
// backend can use saddr-form global ops instead of 64-bit VGPR address adds.
// ---------------------------------------------------------------------------
__global__ __launch_bounds__(128)
void k_main(const float* __restrict__ inputs,
            const float* __restrict__ Wlin,
            const float* __restrict__ blin,
            const float* __restrict__ Ww,
            const float* __restrict__ bw,
            const float* __restrict__ mean,
            const int*   __restrict__ idx,
            const int*   __restrict__ activate_p,
            float*       __restrict__ out,          // [N, 256]
            unsigned int* __restrict__ seg,         // [NVOX, 128] as u32 bits
            int nrows)
{
    const unsigned lane = threadIdx.x & 31u;
    const unsigned wave = threadIdx.x >> 5;
    const unsigned hi   = lane >> 4;      // lane half (0: lanes 0-15, 1: 16-31)
    const unsigned m    = lane & 15u;
    const unsigned row0 = blockIdx.x * 16u;
    if ((int)row0 >= nrows) return;
    const int activate = *activate_p;

    // ---- A fragments (16x32 f16 layout, ISA 7.12.2) -----------------------
    // lane half 0: elems 0-7 = K 0..7,  elems 8-15 = K 16..23
    // lane half 1: elems 0-7 = K 8..15, elems 8-15 = K 24..31
    v16h a0, a1, am;
    {
        const float* arow = inputs + (size_t)(row0 + m) * IN_CH;
        const unsigned kb = hi * 8u;
        #pragma unroll
        for (int i = 0; i < 8; ++i) {
            a0[i]     = (_Float16)arow[kb + i];
            a0[i + 8] = (_Float16)arow[kb + 16 + i];
            a1[i]     = (_Float16)arow[kb + 32 + i];
            a1[i + 8] = (_Float16)arow[kb + 48 + i];
        }
        #pragma unroll
        for (int i = 0; i < 16; ++i) am[i] = (_Float16)0.0f;
        if (hi == 0) {                       // gate GEMM: only K=0..5 nonzero
            const float* mrow = mean + (size_t)(row0 + m) * 6;
            #pragma unroll
            for (int i = 0; i < 6; ++i) am[i] = (_Float16)mrow[i];
        }
    }

    // ---- two 16x16 column tiles per wave ----------------------------------
    #pragma unroll
    for (unsigned t = 0; t < 2; ++t) {
        const unsigned n = wave * 32u + t * 16u + m;   // output column of lane

        // B fragments (32x16): lane half 0 holds K 0..7,16..23; half 1 holds K 8..15,24..31
        v16h b0, b1, bg;
        {
            const float* wrow = Wlin + (size_t)n * IN_CH;
            const unsigned kb = hi * 8u;
            #pragma unroll
            for (int i = 0; i < 8; ++i) {
                b0[i]     = (_Float16)wrow[kb + i];
                b0[i + 8] = (_Float16)wrow[kb + 16 + i];
                b1[i]     = (_Float16)wrow[kb + 32 + i];
                b1[i + 8] = (_Float16)wrow[kb + 48 + i];
            }
            #pragma unroll
            for (int i = 0; i < 16; ++i) bg[i] = (_Float16)0.0f;
            if (hi == 0) {
                const float* wwrow = Ww + (size_t)n * 6;
                #pragma unroll
                for (int i = 0; i < 6; ++i) bg[i] = (_Float16)wwrow[i];
            }
        }

        // bias broadcast into accumulators: C(M,N) = bias[N], same for all r
        v8f accx, accw;
        {
            const float bl = blin[n];
            const float bv = bw[n];
            #pragma unroll
            for (int i = 0; i < 8; ++i) { accx[i] = bl; accw[i] = bv; }
        }

        accx = __builtin_amdgcn_wmma_f32_16x16x32_f16(false, a0, false, b0,
                                                      (short)0, accx, false, false);
        accx = __builtin_amdgcn_wmma_f32_16x16x32_f16(false, a1, false, b1,
                                                      (short)0, accx, false, false);
        accw = __builtin_amdgcn_wmma_f32_16x16x32_f16(false, am, false, bg,
                                                      (short)0, accw, false, false);

        // D layout: VGPR r, this lane -> row = row0 + r + 8*hi, col = n
        #pragma unroll
        for (unsigned r = 0; r < 8; ++r) {
            float xv = accx[r];
            float wv = accw[r];
            if (activate) { xv = fmaxf(xv, 0.0f); wv = fmaxf(wv, 0.0f); }
            const float x = xv * wv;
            const unsigned row = row0 + r + hi * 8u;

            // out[row*256 + n] : 32-bit byte offset (max 512 MB), NT store
            float* outp = (float*)((char*)out + (row * (2u * UNITS * 4u) + n * 4u));
            __builtin_nontemporal_store(x, outp);

            // seg[idx[row]*128 + n] : 32-bit byte offset (max 10.24 MB)
            const unsigned v = (unsigned)idx[row];
            unsigned int* segp =
                (unsigned int*)((char*)seg + (v * (UNITS * 4u) + n * 4u));
            // non-negative f32 orders identically to its u32 bit pattern
            atomicMax(segp, __float_as_uint(x));
        }
    }
}

// ---------------------------------------------------------------------------
// Kernel 3: out[:, 128:256] = seg[idx[row]]   (float4 per thread)
// seg loads stay cached (each voxel row is re-read ~25x); out stores are NT.
// ---------------------------------------------------------------------------
__global__ __launch_bounds__(256)
void k_gather(const int*  __restrict__ idx,
              const unsigned int* __restrict__ seg,  // [NVOX, 128] u32 bits
              float*       __restrict__ out,         // [N, 256]
              int n)                                 // n = N * 32 (uint4 units)
{
    int tid = blockIdx.x * blockDim.x + threadIdx.x;
    if (tid >= n) return;
    const unsigned row = (unsigned)tid >> 5;      // 32 uint4 = 128 floats/row
    const unsigned q   = (unsigned)tid & 31u;

    const unsigned v = (unsigned)idx[row];
    const v4u* sp = (const v4u*)((const char*)seg + (v * (UNITS * 4u) + q * 16u));
    v4u val = *sp;

    // second half of output row: byte offset row*1024 + 512 + q*16 (fits u32)
    v4u* op = (v4u*)((char*)out + (row * (2u * UNITS * 4u) + (UNITS * 4u) + q * 16u));
    __builtin_nontemporal_store(val, op);
}

// ---------------------------------------------------------------------------
extern "C" void kernel_launch(void* const* d_in, const int* in_sizes, int n_in,
                              void* d_out, int out_size, void* d_ws, size_t ws_size,
                              hipStream_t stream) {
    const float* inputs = (const float*)d_in[0];
    const float* Wlin   = (const float*)d_in[1];
    const float* blin   = (const float*)d_in[2];
    const float* Ww     = (const float*)d_in[3];
    const float* bwv    = (const float*)d_in[4];
    const float* mean   = (const float*)d_in[5];
    const int*   idx    = (const int*)d_in[6];
    const int*   act    = (const int*)d_in[7];
    float*       out    = (float*)d_out;
    unsigned int* seg   = (unsigned int*)d_ws;

    const int nrows = in_sizes[0] / IN_CH;        // 500000

    // 1) zero segment-max workspace
    {
        const int n4 = NVOX * UNITS / 4;
        k_seg_init<<<(n4 + 255) / 256, 256, 0, stream>>>((v4u*)seg, n4);
    }
    // 2) fused GEMM + gate + store + segment-max
    {
        const int nblocks = (nrows + 15) / 16;
        k_main<<<nblocks, 128, 0, stream>>>(inputs, Wlin, blin, Ww, bwv, mean,
                                            idx, act, out, seg, nrows);
    }
    // 3) gather into second half of output
    {
        const int n = nrows * (UNITS / 4);
        k_gather<<<(n + 255) / 256, 256, 0, stream>>>(idx, seg, out, n);
    }
}
